// Attention_75411035783199
// MI455X (gfx1250) — matrix operand order repaired and checked by reading, not compiled, not run on hardware
//
#include <hip/hip_runtime.h>

// ---- problem constants (B=8, T=512, C=2048, H=16, hd=128, 3C=6144) ----
#define B_   8
#define T_   512
#define C_   2048
#define H_   16
#define HD_  128
#define F3C_ 6144
#define M_   (B_ * T_)        // 4096 rows of x / qkv / y / out

// ---- block-GEMM tiling ----
#define BM 64
#define BN 128
#define BK 32

typedef __attribute__((ext_vector_type(16))) __bf16 v16bf;
typedef __attribute__((ext_vector_type(8)))  float  v8f;
typedef unsigned int v4u __attribute__((ext_vector_type(4)));
typedef int          v8i __attribute__((ext_vector_type(8)));
typedef int          v4i __attribute__((ext_vector_type(4)));

union FragBF {
    v16bf v;
    uint4 u[2];
};

static __device__ __forceinline__ v8f wmma_bf16(const FragBF& a, const FragBF& b, v8f c) {
    return __builtin_amdgcn_wmma_f32_16x16x32_bf16(
        /*neg_a=*/false, a.v, /*neg_b=*/false, b.v,
        /*c_mod=*/(short)0, c, /*reuse_a=*/false, /*reuse_b=*/false);
}

// Low 32 bits of a generic shared pointer = workgroup-relative LDS byte address
// (flat LDS aperture).
static __device__ __forceinline__ unsigned lds_addr32(const void* p) {
    return (unsigned)(unsigned long long)p;
}

// CDNA5 async copy global->LDS, per-lane 16B, tracked by ASYNCcnt.
static __device__ __forceinline__ void async_ld_b128(const __bf16* g, __bf16* l) {
    asm volatile("global_load_async_to_lds_b128 %0, %1, off"
                 :: "v"(lds_addr32(l)), "v"(g) : "memory");
}

// ---------------------------------------------------------------------------
// Tensor Data Mover: 2D tile load global->LDS via D# descriptor (TENSORcnt).
// D# packing per CDNA5 ISA 8.3/8.4:
//  g0: [1:0]count=1 | [63:32]lds_addr | [120:64]global_addr | [127:126]type=2
//  g1: [17:16]data_size=1(2B) | [79:48]tensor_dim0 | [111:80]tensor_dim1
//      [127:112]tile_dim0 | [143:128]tile_dim1 | [207:160]tensor_dim0_stride
// Tile written to LDS packed row-major [tile_d1][tile_d0].
// This toolchain exposes the 6-arg builtin: (v4u, v8i, v4i, v4i, v8i, i32).
// ---------------------------------------------------------------------------
static __device__ __forceinline__ void tdm_load_2d(
    const void* gsrc, void* ldst,
    unsigned tensor_d0, unsigned tensor_d1,
    unsigned tile_d0, unsigned tile_d1,
    unsigned stride_d0)
{
    const unsigned long long ga = (unsigned long long)gsrc;
    v4u g0;
    g0[0] = 1u;                                    // count=1 (valid user D#)
    g0[1] = lds_addr32(ldst);                      // lds_addr (bytes)
    g0[2] = (unsigned)(ga & 0xffffffffu);          // global_addr[31:0]
    g0[3] = (unsigned)((ga >> 32) & 0x01ffffffu)   // global_addr[56:32]
          | (2u << 30);                            // type=2 ("image")
    v8i g1;
    g1[0] = (int)(1u << 16);                       // data_size=1 -> 2 bytes
    g1[1] = (int)((tensor_d0 & 0xffffu) << 16);                       // dim0 lo
    g1[2] = (int)(((tensor_d0 >> 16) & 0xffffu) |
                  ((tensor_d1 & 0xffffu) << 16));                     // dim0 hi | dim1 lo
    g1[3] = (int)(((tensor_d1 >> 16) & 0xffffu) |
                  ((tile_d0 & 0xffffu) << 16));                       // dim1 hi | tile0
    g1[4] = (int)(tile_d1 & 0xffffu);              // tile_dim1 (tile_dim2=0)
    g1[5] = (int)stride_d0;                        // tensor_dim0_stride[31:0]
    g1[6] = 0;                                     // stride0 hi | stride1 lo
    g1[7] = 0;
    const v4i z4 = {0, 0, 0, 0};                   // 2D: groups 2/3 unused
    const v8i z8 = {0, 0, 0, 0, 0, 0, 0, 0};
    __builtin_amdgcn_tensor_load_to_lds(g0, g1, z4, z4, z8, 0);
}

// ---------------------------------------------------------------------------
// fp32 -> bf16 conversion, 4 elements per thread-iteration
// ---------------------------------------------------------------------------
__global__ void cvt_kernel(const float* __restrict__ src, __bf16* __restrict__ dst, int n4) {
    int i = blockIdx.x * blockDim.x + threadIdx.x;
    const int stride = gridDim.x * blockDim.x;
    for (; i < n4; i += stride) {
        float4 f = ((const float4*)src)[i];
        union { __bf16 h[4]; uint2 u; } pk;
        pk.h[0] = (__bf16)f.x; pk.h[1] = (__bf16)f.y;
        pk.h[2] = (__bf16)f.z; pk.h[3] = (__bf16)f.w;
        ((uint2*)dst)[i] = pk.u;
    }
}

// ---------------------------------------------------------------------------
// Double-buffered block GEMM core, CDNA5 data movers:
//   A tile (64x32)  : per-lane global_load_async_to_lds_b128 (ASYNCcnt)
//   B tile (128x32) : one TDM tensor_load_to_lds descriptor  (TENSORcnt,
//                     wave 0 only -- TDM ignores EXEC, must be branch-guarded)
// 256 threads = 8 waves as 2(M) x 4(N); each wave computes 32x32 (4 WMMA/K32).
// Pipeline: issue tile k+1 -> s_wait_asynccnt 1 / s_wait_tensorcnt 1
//           (batch k done, both counters retire in order) -> barrier
//           -> LDS fragments -> 4x v_wmma -> barrier.
// ---------------------------------------------------------------------------
static __device__ __forceinline__ void issue_tile(
    const __bf16* __restrict__ Ag, const __bf16* __restrict__ Bg,
    int K, int Brows, int mblock, int nblock, int k0,
    __bf16* a_lds, __bf16* b_lds)
{
    const int tid   = threadIdx.x;
    const int row   = tid >> 2;            // 0..63
    const int chunk = (tid & 3) * 8;       // 0,8,16,24 (bf16 elems; 16B)
    async_ld_b128(Ag + (size_t)(mblock + row) * K + k0 + chunk,
                  a_lds + row * BK + chunk);
    if (tid < 32) {
        tdm_load_2d(Bg + (size_t)nblock * K + k0, b_lds,
                    (unsigned)K, (unsigned)Brows, BK, BN, (unsigned)K);
    }
}

static __device__ __forceinline__ void gemm_core(
    const __bf16* __restrict__ Ag, const __bf16* __restrict__ Bg,
    int K, int Brows, int mblock, int nblock,
    __bf16* ldsA, __bf16* ldsB, v8f acc[2][2])
{
    const int lane = threadIdx.x & 31;
    const int w    = threadIdx.x >> 5;
    const int wm   = (w >> 2) * 32;        // 0,32
    const int wn   = (w & 3) * 32;         // 0,32,64,96
    const int lr   = lane & 15;
    const int lg   = lane >> 4;
    const int KT   = K / BK;

    issue_tile(Ag, Bg, K, Brows, mblock, nblock, 0, ldsA, ldsB);

    for (int kt = 0; kt < KT; ++kt) {
        const int cur = kt & 1;
        if (kt + 1 < KT) {
            issue_tile(Ag, Bg, K, Brows, mblock, nblock, (kt + 1) * BK,
                       ldsA + (cur ^ 1) * (BM * BK),
                       ldsB + (cur ^ 1) * (BN * BK));
            asm volatile("s_wait_asynccnt 0x1" ::: "memory");
            if (threadIdx.x < 32) __builtin_amdgcn_s_wait_tensorcnt((short)1);
        } else {
            asm volatile("s_wait_asynccnt 0x0" ::: "memory");
            if (threadIdx.x < 32) __builtin_amdgcn_s_wait_tensorcnt((short)0);
        }
        __syncthreads();

        const __bf16* ab = ldsA + cur * (BM * BK);
        const __bf16* bb = ldsB + cur * (BN * BK);
        FragBF af[2], bfr[2];
#pragma unroll
        for (int mi = 0; mi < 2; ++mi) {
            const __bf16* ar = ab + (wm + mi * 16 + lr) * BK;
            af[mi].u[0] = *(const uint4*)(ar + lg * 8);
            af[mi].u[1] = *(const uint4*)(ar + 16 + lg * 8);
        }
#pragma unroll
        for (int ni = 0; ni < 2; ++ni) {
            const __bf16* br = bb + (wn + ni * 16 + lr) * BK + lg * 8;
            bfr[ni].u[0] = *(const uint4*)(br);
            bfr[ni].u[1] = *(const uint4*)(br + 16);
        }
#pragma unroll
        for (int mi = 0; mi < 2; ++mi)
#pragma unroll
            for (int ni = 0; ni < 2; ++ni)
                acc[mi][ni] = wmma_bf16(af[mi], bfr[ni], acc[mi][ni]);

        __syncthreads();   // protect buffer `cur` before it is re-issued
    }
}

// ---------------------------------------------------------------------------
// QKV projection + RoPE + scatter.
//   q,k  -> [B,H,T,HD] bf16 ; v -> [B,H,HD,T] bf16 (transposed for PV B-frags)
// ---------------------------------------------------------------------------
__global__ __launch_bounds__(256) void qkv_rope_kernel(
    const __bf16* __restrict__ xb,    // [4096, 2048]
    const __bf16* __restrict__ wb,    // [6144, 2048]
    __bf16* __restrict__ qb,
    __bf16* __restrict__ kb,
    __bf16* __restrict__ vtb)
{
    __shared__ __align__(16) __bf16 ldsA[2 * BM * BK];   //  8 KB
    __shared__ __align__(16) __bf16 ldsB[2 * BN * BK];   // 16 KB
    const int nblocks = F3C_ / BN;                        // 48
    const int mblock  = (blockIdx.x / nblocks) * BM;
    const int nblock  = (blockIdx.x % nblocks) * BN;

    v8f acc[2][2] = {};
    gemm_core(xb, wb, C_, F3C_, mblock, nblock, ldsA, ldsB, acc);

    const int lane = threadIdx.x & 31;
    const int w    = threadIdx.x >> 5;
    const int wm   = (w >> 2) * 32;
    const int wn   = (w & 3) * 32;
    const int lr   = lane & 15;
    const int lg   = lane >> 4;

#pragma unroll
    for (int mi = 0; mi < 2; ++mi)
#pragma unroll
    for (int ni = 0; ni < 2; ++ni) {
        const int col  = nblock + wn + ni * 16 + lr;
        const int sidx = col / C_;          // 0=q 1=k 2=v
        const int rem  = col % C_;
        const int h    = rem >> 7;
        const int dd   = rem & (HD_ - 1);
#pragma unroll
        for (int r = 0; r < 8; ++r) {
            const int m = mblock + wm + mi * 16 + r + lg * 8;
            const int b = m >> 9;           // / T_
            const int t = m & (T_ - 1);
            float val = acc[mi][ni][r];
            // RoPE partner of dim d is d^8 -> same 16-lane half, lane^8
            float partner = __shfl_xor(val, 8, 32);
            if (sidx < 2 && dd < 16) {
                const float fi  = (float)(dd & 7);
                const float fr  = __powf(10000.0f, -fi * 0.125f);
                const float ang = (float)t * fr;
                const float cs = __cosf(ang), sn = __sinf(ang);
                val = (dd < 8) ? (val * cs - partner * sn)
                               : (val * cs + partner * sn);
            }
            const int bh = b * H_ + h;
            if (sidx == 0)      qb[(size_t)(bh * T_ + t) * HD_ + dd] = (__bf16)val;
            else if (sidx == 1) kb[(size_t)(bh * T_ + t) * HD_ + dd] = (__bf16)val;
            else                vtb[(size_t)(bh * HD_ + dd) * T_ + t] = (__bf16)val;
        }
    }
}

// ---------------------------------------------------------------------------
// Flash attention: 1 wave per (b,h, 16-query tile). 32 keys per iteration.
//   scores: 8x v_wmma (two 16x16 tiles, K=128 chained as 4xK32)
//   softmax: online, half-wave butterfly reductions
//   PV:     P through wave-private LDS (s_wait_dscnt), 8x v_wmma
// ---------------------------------------------------------------------------
__global__ __launch_bounds__(128) void attn_kernel(
    const __bf16* __restrict__ qb,
    const __bf16* __restrict__ kbuf,
    const __bf16* __restrict__ vtb,
    __bf16* __restrict__ yb)          // [4096, 2048] row-major
{
    __shared__ __align__(16) __bf16 pbuf[4][16 * 32];
    const int lane  = threadIdx.x & 31;
    const int w     = threadIdx.x >> 5;
    const int wid   = blockIdx.x * 4 + w;
    const int qt    = wid & 31;         // 32 query tiles per (b,h)
    const int bh    = wid >> 5;         // 0..127
    const int qbase = qt * 16;
    const int lr = lane & 15;
    const int lg = lane >> 4;
    const int b  = bh >> 4;
    const int h  = bh & 15;

    FragBF qf[4];
    const __bf16* qrow = qb + (size_t)(bh * T_ + qbase + lr) * HD_;
#pragma unroll
    for (int kc = 0; kc < 4; ++kc) {
        qf[kc].u[0] = *(const uint4*)(qrow + kc * 32 + lg * 8);
        qf[kc].u[1] = *(const uint4*)(qrow + kc * 32 + 16 + lg * 8);
    }

    v8f o[8] = {};
    float mrow[8], lrow[8];
#pragma unroll
    for (int r = 0; r < 8; ++r) { mrow[r] = -1e30f; lrow[r] = 0.0f; }

    const float scale = 0.08838834764831845f;   // 1/sqrt(128)
    const int iters = (qbase + 16 + 31) >> 5;   // causal: keys < qbase+16
    __bf16* pw = &pbuf[w][0];

    for (int it = 0; it < iters; ++it) {
        const int kb0 = it * 32;

        v8f s[2] = {};
#pragma unroll
        for (int tile = 0; tile < 2; ++tile) {
            const __bf16* krow = kbuf + (size_t)(bh * T_ + kb0 + tile * 16 + lr) * HD_;
#pragma unroll
            for (int kc = 0; kc < 4; ++kc) {
                FragBF kf;
                kf.u[0] = *(const uint4*)(krow + kc * 32 + lg * 8);
                kf.u[1] = *(const uint4*)(krow + kc * 32 + lg * 8 + 16);
                s[tile] = wmma_bf16(qf[kc], kf, s[tile]);
            }
        }

#pragma unroll
        for (int r = 0; r < 8; ++r) {
            const int qr = qbase + r + lg * 8;
            float s0 = s[0][r] * scale;
            float s1 = s[1][r] * scale;
            if (kb0 + lr > qr)      s0 = -1e30f;
            if (kb0 + 16 + lr > qr) s1 = -1e30f;
            float rmax = fmaxf(s0, s1);
#pragma unroll
            for (int off = 8; off >= 1; off >>= 1)
                rmax = fmaxf(rmax, __shfl_xor(rmax, off, 32));
            const float mnew  = fmaxf(mrow[r], rmax);
            const float alpha = __expf(mrow[r] - mnew);
            const float p0 = __expf(s0 - mnew);
            const float p1 = __expf(s1 - mnew);
            float rsum = p0 + p1;
#pragma unroll
            for (int off = 8; off >= 1; off >>= 1)
                rsum += __shfl_xor(rsum, off, 32);
            lrow[r] = lrow[r] * alpha + rsum;
            mrow[r] = mnew;
#pragma unroll
            for (int nc = 0; nc < 8; ++nc) o[nc][r] *= alpha;
            const int prow = r + lg * 8;
            pw[prow * 32 + lr]      = (__bf16)p0;
            pw[prow * 32 + 16 + lr] = (__bf16)p1;
        }

        // Wave-private LDS; DS ops are in-order per wave -> dscnt wait only
        // (no block barrier: causal loop is wave-divergent).
        asm volatile("s_wait_dscnt 0" ::: "memory");

        FragBF pa;
        pa.u[0] = *(const uint4*)(pw + lr * 32 + lg * 8);
        pa.u[1] = *(const uint4*)(pw + lr * 32 + 16 + lg * 8);

#pragma unroll
        for (int nc = 0; nc < 8; ++nc) {
            const __bf16* vrow = vtb + (size_t)(bh * HD_ + nc * 16 + lr) * T_ + kb0 + lg * 8;
            FragBF vf;
            vf.u[0] = *(const uint4*)(vrow);
            vf.u[1] = *(const uint4*)(vrow + 16);
            o[nc] = wmma_bf16(pa, vf, o[nc]);
        }
    }

#pragma unroll
    for (int r = 0; r < 8; ++r) {
        const float inv = 1.0f / lrow[r];
        const int t = qbase + r + lg * 8;
#pragma unroll
        for (int nc = 0; nc < 8; ++nc) {
            const float val = o[nc][r] * inv;
            yb[(size_t)(b * T_ + t) * C_ + h * HD_ + nc * 16 + lr] = (__bf16)val;
        }
    }
}

// ---------------------------------------------------------------------------
// Output projection: out[m,o] = sum_c y[m,c]*Wout[o,c]  (fp32 out)
// ---------------------------------------------------------------------------
__global__ __launch_bounds__(256) void out_gemm_kernel(
    const __bf16* __restrict__ yb,    // [4096, 2048]
    const __bf16* __restrict__ wob,   // [2048, 2048]
    float* __restrict__ out)          // [4096, 2048] fp32
{
    __shared__ __align__(16) __bf16 ldsA[2 * BM * BK];
    __shared__ __align__(16) __bf16 ldsB[2 * BN * BK];
    const int nblocks = C_ / BN;                          // 16
    const int mblock  = (blockIdx.x / nblocks) * BM;
    const int nblock  = (blockIdx.x % nblocks) * BN;

    v8f acc[2][2] = {};
    gemm_core(yb, wob, C_, C_, mblock, nblock, ldsA, ldsB, acc);

    const int lane = threadIdx.x & 31;
    const int w    = threadIdx.x >> 5;
    const int wm   = (w >> 2) * 32;
    const int wn   = (w & 3) * 32;
    const int lr   = lane & 15;
    const int lg   = lane >> 4;

#pragma unroll
    for (int mi = 0; mi < 2; ++mi)
#pragma unroll
    for (int ni = 0; ni < 2; ++ni) {
        const int col = nblock + wn + ni * 16 + lr;
#pragma unroll
        for (int r = 0; r < 8; ++r) {
            const int m = mblock + wm + mi * 16 + r + lg * 8;
            out[(size_t)m * C_ + col] = acc[mi][ni][r];
        }
    }
}

// ---------------------------------------------------------------------------
// Launch
// ---------------------------------------------------------------------------
extern "C" void kernel_launch(void* const* d_in, const int* in_sizes, int n_in,
                              void* d_out, int out_size, void* d_ws, size_t ws_size,
                              hipStream_t stream) {
    const float* x    = (const float*)d_in[0];   // [8,512,2048]
    const float* Wqkv = (const float*)d_in[1];   // [6144,2048]
    const float* Wout = (const float*)d_in[2];   // [2048,2048]
    float* out = (float*)d_out;

    char* ws = (char*)d_ws;
    __bf16* xb  = (__bf16*)(ws + 0);            //  16.78 MB
    __bf16* wqb = (__bf16*)(ws + 16777216);     //  25.17 MB
    __bf16* wob = (__bf16*)(ws + 41943040);     //   8.39 MB
    __bf16* qb  = (__bf16*)(ws + 50331648);     //  16.78 MB
    __bf16* kb  = (__bf16*)(ws + 67108864);     //  16.78 MB
    __bf16* vtb = (__bf16*)(ws + 83886080);     //  16.78 MB (transposed V)
    __bf16* yb  = (__bf16*)(ws + 100663296);    //  16.78 MB

    cvt_kernel<<<2048, 256, 0, stream>>>(x,    xb,  (M_ * C_) / 4);
    cvt_kernel<<<2048, 256, 0, stream>>>(Wqkv, wqb, (F3C_ * C_) / 4);
    cvt_kernel<<<1024, 256, 0, stream>>>(Wout, wob, (C_ * C_) / 4);

    // 64 m-blocks * 48 n-blocks, 256 threads (8 waves) each
    qkv_rope_kernel<<<3072, 256, 0, stream>>>(xb, wqb, qb, kb, vtb);

    // 128 (b,h) * 32 q-tiles = 4096 waves, 4 waves/block
    attn_kernel<<<1024, 128, 0, stream>>>(qb, kb, vtb, yb);

    // 64 m-blocks * 16 n-blocks
    out_gemm_kernel<<<1024, 256, 0, stream>>>(yb, wob, out);
}
